// DeepSeekMoELayer_18425409700063
// MI455X (gfx1250) — compile-verified, matrix operand order checked by reading, not run on hardware
//
#include <hip/hip_runtime.h>
#include <hip/hip_bf16.h>

// Problem constants: B=4, L=2048 -> T=8192 tokens, D=1024, F=512,
// E=16 routed experts (top-2) + 1 shared expert.
#define T_TOK 8192
#define DDIM  1024
#define FDIM  512
#define NEXP  16
#define NSEG  17   // 16 routed + 1 shared segment

typedef __attribute__((ext_vector_type(16))) __bf16        v16bf;
typedef __attribute__((ext_vector_type(8)))  float         v8f;
typedef __attribute__((ext_vector_type(4)))  float         v4f;
typedef __attribute__((ext_vector_type(4)))  unsigned int  v4u;
typedef __attribute__((ext_vector_type(8)))  int           v8i;
typedef __attribute__((ext_vector_type(4)))  int           v4i;

struct alignas(16) U128 { unsigned int a, b, c, d; };
union Frag { v16bf v; unsigned short u[16]; U128 q[2]; };

__device__ inline unsigned short f2bf(float f) {
  unsigned int x = __float_as_uint(f);
  x += 0x7FFFu + ((x >> 16) & 1u);          // round-to-nearest-even
  return (unsigned short)(x >> 16);
}

// ---------------------------------------------------------------------------
// One-time fp32 -> bf16 conversion (HBM-bound; pays the cast once instead of
// once per tile in the WMMA inner loop). 8 elements per thread.
// ---------------------------------------------------------------------------
__global__ void k_cvt_bf16(const float* __restrict__ src,
                           unsigned short* __restrict__ dst, int n8) {
  int i = blockIdx.x * blockDim.x + threadIdx.x;
  if (i >= n8) return;
  const float4* s = (const float4*)src + (size_t)i * 2;
  float4 a = s[0], b = s[1];
  U128 o;
  o.a = (unsigned)f2bf(a.x) | ((unsigned)f2bf(a.y) << 16);
  o.b = (unsigned)f2bf(a.z) | ((unsigned)f2bf(a.w) << 16);
  o.c = (unsigned)f2bf(b.x) | ((unsigned)f2bf(b.y) << 16);
  o.d = (unsigned)f2bf(b.z) | ((unsigned)f2bf(b.w) << 16);
  ((U128*)dst)[i] = o;
}

__global__ void k_zero_counts(int* __restrict__ cnt) {
  int i = threadIdx.x;
  if (i < NEXP) cnt[i] = 0;
  else if (i == NEXP) cnt[NEXP] = T_TOK;    // shared segment covers all tokens
}

// ---------------------------------------------------------------------------
// Router: one wave32 per token. Lanes 0-15 = experts, the two 16-lane halves
// split D. Softmax + top-2 via group-of-16 shuffle reductions. Builds
// per-expert gather lists (int atomics only; per-row math is list-order
// independent, so the final output stays deterministic).
// ---------------------------------------------------------------------------
__global__ void __launch_bounds__(256) k_router(
    const float* __restrict__ x, const float* __restrict__ gw,
    const float* __restrict__ bias, float* __restrict__ probs,
    int* __restrict__ cnt, int* __restrict__ tok, float* __restrict__ tw) {
  int t    = (blockIdx.x * blockDim.x + threadIdx.x) >> 5;
  int lane = threadIdx.x & 31;
  if (t >= T_TOK) return;
  int e = lane & 15, h = lane >> 4;

  const float4* x4 = (const float4*)(x  + (size_t)t * DDIM + h * (DDIM / 2));
  const float4* g4 = (const float4*)(gw + (size_t)e * DDIM + h * (DDIM / 2));
  float acc = 0.f;
#pragma unroll 4
  for (int i = 0; i < DDIM / 8; ++i) {
    float4 a = x4[i], b = g4[i];
    acc += a.x * b.x + a.y * b.y + a.z * b.z + a.w * b.w;
  }
  float logit = acc + __shfl_xor(acc, 16) + bias[e];

  float mx = logit;
  for (int m = 8; m; m >>= 1) mx = fmaxf(mx, __shfl_xor(mx, m));
  float pe = __expf(logit - mx);
  float sm = pe;
  for (int m = 8; m; m >>= 1) sm += __shfl_xor(sm, m);
  float pr = pe * __builtin_amdgcn_rcpf(sm);
  if (lane < 16) probs[(size_t)t * NEXP + e] = pr;

  float m1 = pr; int i1 = e;                 // argmax (tie -> lowest index)
  for (int m = 8; m; m >>= 1) {
    float om = __shfl_xor(m1, m); int oi = __shfl_xor(i1, m);
    if (om > m1 || (om == m1 && oi < i1)) { m1 = om; i1 = oi; }
  }
  float p2 = (e == i1) ? -1.f : pr;          // second max
  float m2 = p2; int i2 = e;
  for (int m = 8; m; m >>= 1) {
    float om = __shfl_xor(m2, m); int oi = __shfl_xor(i2, m);
    if (om > m2 || (om == m2 && oi < i2)) { m2 = om; i2 = oi; }
  }
  float inv = __builtin_amdgcn_rcpf(m1 + m2);

  if (lane == 0) {
    int p = atomicAdd(cnt + i1, 1);
    tok[(size_t)i1 * T_TOK + p] = t;                  // slot 0
    tw [(size_t)i1 * T_TOK + p] = m1 * inv;
    p = atomicAdd(cnt + i2, 1);
    tok[(size_t)i2 * T_TOK + p] = t | (1 << 16);      // slot 1
    tw [(size_t)i2 * T_TOK + p] = m2 * inv;
    tok[(size_t)NEXP * T_TOK + t] = t;                // shared identity list
    tw [(size_t)NEXP * T_TOK + t] = 1.f;
  }
}

// ---------------------------------------------------------------------------
// Fused grouped SwiGLU FFN: one 16-token tile x one segment per block.
// x-tile gather uses the Tensor Data Mover in gather mode (16 x 16-bit row
// indices, 2-byte elements) -> one async 32 KB LDS fill per block.
// Phase 1: h = silu(x@w1^T) * (x@w3^T)  (bf16 WMMA, K=D=1024)
// Phase 2: o = h @ w2^T                 (bf16 WMMA, K=F=512)
// ---------------------------------------------------------------------------
__global__ void __launch_bounds__(256) k_moe_ffn(
    const unsigned short* __restrict__ xbf,
    const unsigned short* __restrict__ w1bf,   // [17][F][D] bf16
    const unsigned short* __restrict__ w2bf,   // [17][D][F] bf16
    const unsigned short* __restrict__ w3bf,   // [17][F][D] bf16
    const int* __restrict__ cnt, const int* __restrict__ tok,
    const float* __restrict__ tw,
    float* __restrict__ routed, float* __restrict__ y) {
  const int e    = blockIdx.y;
  const int rows = cnt[e];
  const int base = blockIdx.x * 16;
  if (base >= rows) return;                 // tile-skip for sparse experts

  __shared__ unsigned short xs[16][DDIM];   // 32 KB gathered x tile (bf16)
  __shared__ unsigned short hs[16][FDIM];   // 16 KB intermediate (bf16)
  __shared__ int tokL[16]; __shared__ int slotL[16];
  __shared__ float wL[16]; __shared__ int okL[16];

  const int tid = threadIdx.x;
  if (tid < 16) {
    int r = base + tid;
    if (r < rows) {
      int en = tok[(size_t)e * T_TOK + r];
      tokL[tid] = en & 0xFFFF; slotL[tid] = (en >> 16) & 1;
      wL[tid] = tw[(size_t)e * T_TOK + r]; okL[tid] = 1;
    } else { tokL[tid] = 0; slotL[tid] = 0; wL[tid] = 0.f; okL[tid] = 0; }
  }
  __syncthreads();

#if defined(__HIP_DEVICE_COMPILE__)
  // ---- TDM gather: 16 token rows (bf16, 2 KB each) -> LDS ----
  if (tid < 32) {                            // wave 0 only (TDM ignores EXEC)
    unsigned iw[8];
#pragma unroll
    for (int j = 0; j < 8; ++j)
      iw[j] = (unsigned)tokL[2 * j] | ((unsigned)tokL[2 * j + 1] << 16);

    unsigned long long ga = (unsigned long long)(size_t)xbf;
    unsigned ldsoff = (unsigned)(size_t)&xs[0][0];

    v4u g0; v8i g1; v4i g2, g3;
    g0[0] = 0x80000001u;                      // count=1, gather_mode=1, 16b idx
    g0[1] = ldsoff;                           // lds_addr
    g0[2] = (unsigned)(ga & 0xFFFFFFFFu);     // global_addr[31:0]
    g0[3] = (unsigned)((ga >> 32) & 0x1FFFFFFu) | (2u << 30);  // addr hi | type=2
    g1[0] = 0x00010000;                       // data_size = 2 bytes
    g1[1] = (int)(0x0400u << 16);             // tensor_dim0 = 1024
    g1[2] = (int)(0x2000u << 16);             // tensor_dim1 = 8192
    g1[3] = (int)(0x0400u << 16);             // tile_dim0 = 1024
    g1[4] = 16;                               // tile_dim1 = #gather indices
    g1[5] = 1024;                             // tensor_dim0_stride
    g1[6] = 0; g1[7] = 0;
#pragma unroll
    for (int j = 0; j < 4; ++j) {
      g2[j] = __builtin_amdgcn_readfirstlane((int)iw[j]);
      g3[j] = __builtin_amdgcn_readfirstlane((int)iw[j + 4]);
    }
#if __clang_major__ >= 23
    v8i g4 = {0, 0, 0, 0, 0, 0, 0, 0};
    __builtin_amdgcn_tensor_load_to_lds(g0, g1, g2, g3, g4, 0);
#else
    __builtin_amdgcn_tensor_load_to_lds(g0, g1, g2, g3, 0);
#endif
    __builtin_amdgcn_s_wait_tensorcnt(0);
  }
#else
  // host-parse fallback (never executed on device)
  {
    int r = tid >> 4, c0 = (tid & 15) * 64;
    const U128* src = (const U128*)(xbf + (size_t)tokL[r] * DDIM + c0);
    U128* dstp = (U128*)&xs[r][c0];
    for (int i = 0; i < 8; ++i) dstp[i] = src[i];
  }
#endif
  __syncthreads();

  const unsigned short* W1 = w1bf + (size_t)e * FDIM * DDIM;
  const unsigned short* W3 = w3bf + (size_t)e * FDIM * DDIM;
  const unsigned short* W2 = w2bf + (size_t)e * DDIM * FDIM;

  const int wave = tid >> 5, lane = tid & 31;
  const int ln = lane & 15, lh = lane >> 4, kh = lh * 8;

  // ---- phase 1: 32 N-tiles over F, 4 per wave, K = D = 1024 ----
  for (int nt = wave * 4; nt < wave * 4 + 4; ++nt) {
    const int n = nt * 16 + ln;
    const unsigned short* r1 = W1 + (size_t)n * DDIM;
    const unsigned short* r3 = W3 + (size_t)n * DDIM;
    v8f acc1 = {}; v8f acc3 = {};
    for (int kb = 0; kb < DDIM; kb += 32) {
      __builtin_prefetch(r1 + kb + 128, 0, 1);
      __builtin_prefetch(r3 + kb + 128, 0, 1);
      Frag a, b1, b3;
      a.q[0]  = *(const U128*)&xs[ln][kb + kh];
      a.q[1]  = *(const U128*)&xs[ln][kb + 16 + kh];
      b1.q[0] = *(const U128*)(r1 + kb + kh);
      b1.q[1] = *(const U128*)(r1 + kb + 16 + kh);
      b3.q[0] = *(const U128*)(r3 + kb + kh);
      b3.q[1] = *(const U128*)(r3 + kb + 16 + kh);
      acc1 = __builtin_amdgcn_wmma_f32_16x16x32_bf16(false, a.v, false, b1.v, (short)0, acc1, false, false);
      acc3 = __builtin_amdgcn_wmma_f32_16x16x32_bf16(false, a.v, false, b3.v, (short)0, acc3, false, false);
    }
#pragma unroll
    for (int i = 0; i < 8; ++i) {
      int M = i + lh * 8;
      float c1 = acc1[i], c3 = acc3[i];
      float hv = c1 * __builtin_amdgcn_rcpf(1.f + __expf(-c1)) * c3;  // silu*gate
      hs[M][nt * 16 + ln] = f2bf(hv);
    }
  }
  __syncthreads();

  // ---- phase 2: 64 N-tiles over D, 8 per wave, K = F = 512 ----
  for (int nt = wave * 8; nt < wave * 8 + 8; ++nt) {
    const int n = nt * 16 + ln;
    const unsigned short* r2 = W2 + (size_t)n * FDIM;
    v8f acc = {};
    for (int kb = 0; kb < FDIM; kb += 32) {
      __builtin_prefetch(r2 + kb + 128, 0, 1);
      Frag a, b;
      a.q[0] = *(const U128*)&hs[ln][kb + kh];
      a.q[1] = *(const U128*)&hs[ln][kb + 16 + kh];
      b.q[0] = *(const U128*)(r2 + kb + kh);
      b.q[1] = *(const U128*)(r2 + kb + 16 + kh);
      acc = __builtin_amdgcn_wmma_f32_16x16x32_bf16(false, a.v, false, b.v, (short)0, acc, false, false);
    }
#pragma unroll
    for (int i = 0; i < 8; ++i) {
      int M = i + lh * 8;
      if (okL[M]) {
        int t = tokL[M];
        float val = acc[i];
        if (e == NEXP) y[(size_t)t * DDIM + n] = val;               // shared, w=1
        else __builtin_nontemporal_store(val * wL[M],               // streamed
               routed + ((size_t)slotL[M] * T_TOK + t) * DDIM + n); // unique (t,slot)
      }
    }
  }
}

__global__ void k_combine(float* __restrict__ y, const float* __restrict__ routed) {
  const int n4 = T_TOK * DDIM / 4;
  int i = blockIdx.x * blockDim.x + threadIdx.x;
  if (i >= n4) return;
  v4f* y4 = (v4f*)y;
  const v4f* r0 = (const v4f*)routed;
  const v4f* r1 = r0 + n4;
  v4f a = y4[i];
  v4f b = __builtin_nontemporal_load(r0 + i);
  v4f c = __builtin_nontemporal_load(r1 + i);
  a = a + b + c;
  y4[i] = a;
}

extern "C" void kernel_launch(void* const* d_in, const int* in_sizes, int n_in,
                              void* d_out, int out_size, void* d_ws, size_t ws_size,
                              hipStream_t stream) {
  const float* x    = (const float*)d_in[0];
  const float* gw   = (const float*)d_in[1];
  const float* bias = (const float*)d_in[2];
  const float* sw1  = (const float*)d_in[3];
  const float* sw2  = (const float*)d_in[4];
  const float* sw3  = (const float*)d_in[5];
  const float* ew1  = (const float*)d_in[6];
  const float* ew2  = (const float*)d_in[7];
  const float* ew3  = (const float*)d_in[8];

  float* y     = (float*)d_out;                       // [T, D]
  float* probs = y + (size_t)T_TOK * DDIM;            // [T, E]

  const size_t FD = (size_t)FDIM * DDIM;              // 524288
  // Workspace: counters | lists | bf16 copies (x, w1, w2, w3) | routed[2][T][D]
  char*  ws  = (char*)d_ws;
  int*   cnt = (int*)ws;                                              // 64 B
  int*   tok = (int*)(ws + 64);                                       // 17*T ints
  float* tw  = (float*)(ws + 64 + (size_t)NSEG * T_TOK * 4);          // 17*T floats
  size_t off = ((size_t)64 + (size_t)2 * NSEG * T_TOK * 4 + 255) & ~(size_t)255;
  unsigned short* xbf  = (unsigned short*)(ws + off); off += (size_t)T_TOK * DDIM * 2;
  unsigned short* w1bf = (unsigned short*)(ws + off); off += NSEG * FD * 2;
  unsigned short* w2bf = (unsigned short*)(ws + off); off += NSEG * FD * 2;
  unsigned short* w3bf = (unsigned short*)(ws + off); off += NSEG * FD * 2;
  off = (off + 255) & ~(size_t)255;
  float* routed = (float*)(ws + off);                                 // 64 MB

  // One-time bf16 staging (memory-bound, weights become L2-resident bf16).
  auto cvt = [&](const float* s, unsigned short* d, size_t n) {
    int n8 = (int)(n / 8);
    k_cvt_bf16<<<(n8 + 255) / 256, 256, 0, stream>>>(s, d, n8);
  };
  cvt(x,   xbf,  (size_t)T_TOK * DDIM);
  cvt(ew1, w1bf, 16 * FD);  cvt(sw1, w1bf + 16 * FD, FD);
  cvt(ew2, w2bf, 16 * FD);  cvt(sw2, w2bf + 16 * FD, FD);
  cvt(ew3, w3bf, 16 * FD);  cvt(sw3, w3bf + 16 * FD, FD);

  k_zero_counts<<<1, 32, 0, stream>>>(cnt);
  k_router<<<T_TOK / 8, 256, 0, stream>>>(x, gw, bias, probs, cnt, tok, tw);
  k_moe_ffn<<<dim3(T_TOK / 16, NSEG), 256, 0, stream>>>(
      xbf, w1bf, w2bf, w3bf, cnt, tok, tw, routed, y);
  k_combine<<<(T_TOK * DDIM / 4) / 256, 256, 0, stream>>>(y, routed);
}
